// qLinear_5188320494015
// MI455X (gfx1250) — matrix operand order repaired and checked by reading, not compiled, not run on hardware
//
#include <hip/hip_runtime.h>

// qLinear: out[B,M] = x[B,N] @ dequant4(qweight)[N,M] + bias
#define TOK_B 2048
#define NFEAT 4096
#define MFEAT 11008
#define QGS   128

#define BM 128          // tokens per workgroup tile
#define BN 128          // out-features per workgroup tile
#define BK 64           // K rows per LDS stage (inside one quant group)
#define LDSK 72         // padded K stride in halves (144 B, 16B aligned)
#define NSTAGES (NFEAT / BK)

typedef _Float16 v16h __attribute__((ext_vector_type(16)));
typedef _Float16 v8h  __attribute__((ext_vector_type(8)));
typedef float    v8f  __attribute__((ext_vector_type(8)));

// Register-staged prefetch for one K stage.
struct PF {
    float4 xv[8];   // 128x64 f32 activation tile, 8 float4 per thread
    uint4  qv;      // 8x128 packed words: 1 uint4 per thread (4 consecutive m)
    float4 sv;      // group scales for those 4 m
    float4 zv;      // group zeros  for those 4 m
};

__global__ __launch_bounds__(256)
void qlinear_wmma_f16(const float* __restrict__ x,
                      const unsigned int* __restrict__ qw,
                      const float* __restrict__ scales,
                      const float* __restrict__ zeros,
                      const float* __restrict__ bias,
                      float* __restrict__ out)
{
    __shared__ _Float16 At[BM][LDSK];   // At[token][k]   (f16 activations)
    __shared__ _Float16 Bt[BN][LDSK];   // Bt[outcol][k]  (dequantized, transposed)

    const int tid  = threadIdx.x;
    const int lane = tid & 31;          // wave32
    const int wid  = tid >> 5;          // 8 waves
    const int wrow = wid >> 1;          // 0..3 : token direction  (4 * 32 = 128)
    const int wcol = wid & 1;           // 0..1 : out  direction   (2 * 64 = 128)

    const int tokBase = blockIdx.y * BM;
    const int outBase = blockIdx.x * BN;

    const int hi16 = lane >> 4;         // 0: lanes 0-15, 1: lanes 16-31
    const int l15  = lane & 15;

    // per-thread fixed coordinates for the staging phases
    const int qwr = tid >> 5;           // packed word-row 0..7
    const int qm0 = (tid & 31) * 4;     // first of 4 consecutive out-cols

    v8f acc[2][4];
#pragma unroll
    for (int i = 0; i < 2; ++i)
#pragma unroll
        for (int j = 0; j < 4; ++j)
#pragma unroll
            for (int e = 0; e < 8; ++e)
                acc[i][j][e] = 0.0f;

    PF pf;

    // ---- issue global loads for stage 0 ----
    {
        const int k0 = 0;
#pragma unroll
        for (int i = 0; i < 8; ++i) {
            const int idx = tid + 256 * i;
            const int row = idx >> 4;             // 16 float4 per 64-wide row
            const int c4  = idx & 15;
            pf.xv[i] = *(const float4*)(x + (size_t)(tokBase + row) * NFEAT + k0 + c4 * 4);
        }
        const int g = k0 / QGS;
        pf.qv = *(const uint4*)(qw + (size_t)((k0 >> 3) + qwr) * MFEAT + outBase + qm0);
        pf.sv = *(const float4*)(scales + (size_t)g * MFEAT + outBase + qm0);
        pf.zv = *(const float4*)(zeros  + (size_t)g * MFEAT + outBase + qm0);
    }

    for (int stage = 0; stage < NSTAGES; ++stage) {
        const int k0 = stage * BK;

        // ---- drain prefetch registers into LDS (f32->f16 convert + dequant) ----
#pragma unroll
        for (int i = 0; i < 8; ++i) {
            const int idx = tid + 256 * i;
            const int row = idx >> 4;
            const int c4  = idx & 15;
            _Float16* dst = &At[row][c4 * 4];
            dst[0] = (_Float16)pf.xv[i].x;
            dst[1] = (_Float16)pf.xv[i].y;
            dst[2] = (_Float16)pf.xv[i].z;
            dst[3] = (_Float16)pf.xv[i].w;
        }
        {
            const unsigned int qarr[4] = {pf.qv.x, pf.qv.y, pf.qv.z, pf.qv.w};
            const float        sarr[4] = {pf.sv.x, pf.sv.y, pf.sv.z, pf.sv.w};
            const float        zarr[4] = {pf.zv.x, pf.zv.y, pf.zv.z, pf.zv.w};
#pragma unroll
            for (int c = 0; c < 4; ++c) {
                const unsigned int wbits = qarr[c];
                const float s = sarr[c];
                const float z = zarr[c];
                v8h pk;
#pragma unroll
                for (int j = 0; j < 8; ++j) {
                    const float w = ((float)((wbits >> (4 * j)) & 0xFu) - z) * s;
                    pk[j] = (_Float16)w;
                }
                *(v8h*)&Bt[qm0 + c][qwr * 8] = pk;   // 16B aligned LDS store
            }
        }

        __syncthreads();

        // ---- issue next stage's global loads NOW; WMMA below hides them ----
        if (stage + 1 < NSTAGES) {
            const int kn = k0 + BK;
#pragma unroll
            for (int i = 0; i < 8; ++i) {
                const int idx = tid + 256 * i;
                const int row = idx >> 4;
                const int c4  = idx & 15;
                pf.xv[i] = *(const float4*)(x + (size_t)(tokBase + row) * NFEAT + kn + c4 * 4);
            }
            const int g = kn / QGS;
            pf.qv = *(const uint4*)(qw + (size_t)((kn >> 3) + qwr) * MFEAT + outBase + qm0);
            pf.sv = *(const float4*)(scales + (size_t)g * MFEAT + outBase + qm0);
            pf.zv = *(const float4*)(zeros  + (size_t)g * MFEAT + outBase + qm0);
        }

        // ---- compute: two K=32 WMMA steps over this stage ----
#pragma unroll
        for (int ks = 0; ks < BK; ks += 32) {
            // ISA 16-bit A 16x32 layout: lanes 0-15 -> K 0-7 / 16-23,
            //                            lanes 16-31 -> K 8-15 / 24-31.
            v16h afrag[2];
#pragma unroll
            for (int tm = 0; tm < 2; ++tm) {
                const int mrow = wrow * 32 + tm * 16 + l15;
                const _Float16* p = &At[mrow][ks + hi16 * 8];
                const v8h lo = *(const v8h*)p;
                const v8h hi = *(const v8h*)(p + 16);
                v16h a;
#pragma unroll
                for (int e = 0; e < 8; ++e) { a[e] = lo[e]; a[e + 8] = hi[e]; }
                afrag[tm] = a;
            }
            // B: lane group 0 holds K 0-7 / 16-23, group 1 holds K 8-15 / 24-31 (same as A).
#pragma unroll
            for (int tn = 0; tn < 4; ++tn) {
                const int ncol = wcol * 64 + tn * 16 + l15;
                const _Float16* p = &Bt[ncol][ks + hi16 * 8];
                const v8h lo = *(const v8h*)p;
                const v8h hi = *(const v8h*)(p + 16);
                v16h b;
#pragma unroll
                for (int e = 0; e < 8; ++e) { b[e] = lo[e]; b[e + 8] = hi[e]; }
#pragma unroll
                for (int tm = 0; tm < 2; ++tm) {
                    acc[tm][tn] = __builtin_amdgcn_wmma_f32_16x16x32_f16(
                        /*neg_a=*/false, afrag[tm],
                        /*neg_b=*/false, b,
                        /*c_mod=*/(short)0, acc[tm][tn],
                        /*reuse_a=*/false, /*reuse_b=*/false);
                }
            }
        }

        __syncthreads();
    }

    // ---- epilogue: C/D layout -> lanes 0-15 rows 0-7, lanes 16-31 rows 8-15 ----
    // Non-temporal stores: out (90 MB, write-once) must not evict x/qweight
    // (56 MB, reused across the whole grid) from the 192 MB L2.
#pragma unroll
    for (int tm = 0; tm < 2; ++tm) {
#pragma unroll
        for (int tn = 0; tn < 4; ++tn) {
            const int ncol = outBase + wcol * 64 + tn * 16 + l15;
            const float bv = bias[ncol];
#pragma unroll
            for (int r = 0; r < 8; ++r) {
                const int trow = tokBase + wrow * 32 + tm * 16 + hi16 * 8 + r;
                __builtin_nontemporal_store(acc[tm][tn][r] + bv,
                                            out + (size_t)trow * MFEAT + ncol);
            }
        }
    }
}

extern "C" void kernel_launch(void* const* d_in, const int* in_sizes, int n_in,
                              void* d_out, int out_size, void* d_ws, size_t ws_size,
                              hipStream_t stream) {
    (void)in_sizes; (void)n_in; (void)out_size; (void)d_ws; (void)ws_size;
    const float*        x      = (const float*)d_in[0];
    const unsigned int* qw     = (const unsigned int*)d_in[1];
    const float*        scales = (const float*)d_in[2];
    const float*        zeros  = (const float*)d_in[3];
    const float*        bias   = (const float*)d_in[4];
    float*              out    = (float*)d_out;

    dim3 grid(MFEAT / BN, TOK_B / BM);   // 86 x 16
    dim3 block(256);
    qlinear_wmma_f16<<<grid, block, 0, stream>>>(x, qw, scales, zeros, bias, out);
}
